// HardTripletLoss_26594437497378
// MI455X (gfx1250) — matrix operand order repaired and checked by reading, not compiled, not run on hardware
//
#include <hip/hip_runtime.h>
#include <hip/hip_bf16.h>

typedef __attribute__((ext_vector_type(16))) _Float16 v16h;
typedef __attribute__((ext_vector_type(8)))  _Float16 v8h;
typedef __attribute__((ext_vector_type(4)))  _Float16 v4h;
typedef __attribute__((ext_vector_type(8)))  float    v8f;

#define NROWS   4096
#define DIMS    512
#define MARGIN  1.0f

// d_ws layout:
//   [0, 4 MB)                : normalized embeddings, f16, row-major 4096x512
//   [4 MB, 4 MB+16 KB)       : per-row squared norms (f32)
//   [4 MB+16 KB, ... )       : accum[0]=total loss, accum[1]=valid row count
#define WS_NF16_BYTES   (NROWS * DIMS * 2)
#define WS_SQN_OFF      WS_NF16_BYTES
#define WS_ACC_OFF      (WS_NF16_BYTES + NROWS * 4)

// ---------------------------------------------------------------------------
// Kernel 1: L2-normalize each row, cast to f16, store squared norm of the
// normalized row (matches reference: sq_norms computed on normalized x).
// Block 0 also zeroes the global accumulators (re-done on every launch).
// ---------------------------------------------------------------------------
__global__ __launch_bounds__(128)
void tl_normalize(const float* __restrict__ emb,
                  _Float16* __restrict__ nf16,
                  float* __restrict__ sqn,
                  float* __restrict__ accum) {
    const int row = blockIdx.x;
    const int t   = threadIdx.x;            // 0..127, 4 floats each
    const float4 v = ((const float4*)(emb + (size_t)row * DIMS))[t];
    float ssq = v.x * v.x + v.y * v.y + v.z * v.z + v.w * v.w;
    // wave32 reduction
    #pragma unroll
    for (int off = 16; off >= 1; off >>= 1)
        ssq += __shfl_xor(ssq, off, 32);
    __shared__ float sw[4];
    if ((t & 31) == 0) sw[t >> 5] = ssq;
    __syncthreads();
    const float tot  = sw[0] + sw[1] + sw[2] + sw[3];
    const float norm = sqrtf(tot);
    const float inv  = 1.0f / fmaxf(norm, 1e-12f);

    v4h h;
    h[0] = (_Float16)(v.x * inv);
    h[1] = (_Float16)(v.y * inv);
    h[2] = (_Float16)(v.z * inv);
    h[3] = (_Float16)(v.w * inv);
    *(v4h*)(nf16 + (size_t)row * DIMS + t * 4) = h;

    if (t == 0) {
        sqn[row] = tot * inv * inv;         // squared norm of normalized row
        if (row == 0) { accum[0] = 0.0f; accum[1] = 0.0f; }
    }
}

// ---------------------------------------------------------------------------
// Kernel 2: fused Gram (f16 WMMA, f32 acc) + per-row triplet statistics.
// Grid: 256 blocks (16 rows each). Block: 256 threads = 8 waves; wave w owns
// columns [w*512, w*512+512), so the block covers all 4096 columns.
// ---------------------------------------------------------------------------
__global__ __launch_bounds__(256)
void tl_gram_reduce(const _Float16* __restrict__ nf16,
                    const float* __restrict__ sqn,
                    const long long* __restrict__ labels,
                    float* __restrict__ accum) {
    __shared__ int      sLab[NROWS];
    __shared__ float    sSqn[NROWS];
    __shared__ float    sPos[16];
    __shared__ float    sCnt[16];
    __shared__ unsigned sMin[16];

    const int t = threadIdx.x;
    for (int i = t; i < NROWS; i += 256) {
        sLab[i] = (int)labels[i];
        sSqn[i] = sqn[i];
    }
    if (t < 16) { sPos[t] = 0.0f; sCnt[t] = 0.0f; sMin[t] = 0x7f7fffffu; }
    __syncthreads();

    const int mBase = blockIdx.x * 16;
    const int wave  = t >> 5;
    const int lane  = t & 31;
    const int hi    = lane >> 4;            // half-wave selector
    const int ln    = lane & 15;

    // ---- A fragments: 16 rows x K=512, kept in registers (16 x v16h). ----
    // A k order chosen here (contiguous, not ISA table order): lanes 0-15 hold K[k0..k0+15],
    // lanes 16-31 hold K[k0+16..k0+31] (matches B).
    v16h a[16];
    const _Float16* arow = nf16 + (size_t)(mBase + ln) * DIMS;
    #pragma unroll
    for (int kc = 0; kc < 16; ++kc) {
        const _Float16* p = arow + kc * 32 + hi * 16;
        union { v16h v; v8h h[2]; } u;
        u.h[0] = *(const v8h*)(p);          // 16B aligned
        u.h[1] = *(const v8h*)(p + 8);      // 16B aligned
        a[kc] = u.v;
    }

    // Row-side constants for the 8 C rows this lane sees (row = mBase+v+8*hi)
    float rowSqn[8]; int rowLab[8];
    #pragma unroll
    for (int v = 0; v < 8; ++v) {
        const int r = mBase + v + 8 * hi;
        rowSqn[v] = sSqn[r];
        rowLab[v] = sLab[r];
    }

    float psum[8], pcnt[8], nmin[8];
    #pragma unroll
    for (int v = 0; v < 8; ++v) { psum[v] = 0.0f; pcnt[v] = 0.0f; nmin[v] = 3.0e38f; }

    const int jStart = wave * 512;
    for (int jb = 0; jb < 512; jb += 16) {
        const int col = jStart + jb + ln;   // this lane's column
        // B k order chosen here (contiguous, same as A): lane = column, lanes 0-15 hold K[k0..k0+15],
        // lanes 16-31 hold K[k0+16..k0+31] -> one 32B contiguous load.
        const _Float16* brow = nf16 + (size_t)col * DIMS;
        v8f c = {};
        #pragma unroll
        for (int kc = 0; kc < 16; ++kc) {
            const v16h b = *(const v16h*)(brow + kc * 32 + hi * 16); // 32B aligned
            c = __builtin_amdgcn_wmma_f32_16x16x32_f16(
                    false, a[kc], false, b, (short)0, c, false, false);
        }
        const float colSqn = sSqn[col];
        const int   colLab = sLab[col];
        #pragma unroll
        for (int v = 0; v < 8; ++v) {
            const int r = mBase + v + 8 * hi;
            float sq = rowSqn[v] + colSqn - 2.0f * c[v];
            sq = fmaxf(sq, 0.0f);
            const float d = (sq > 0.0f) ? sqrtf(sq) : 0.0f;
            if (col != r) {
                if (rowLab[v] == colLab) { psum[v] += d; pcnt[v] += 1.0f; }
                else                      nmin[v]  = fminf(nmin[v], d);
            }
        }
    }

    // Reduce across the 16 lanes of each half-wave (same rows, distinct cols)
    #pragma unroll
    for (int v = 0; v < 8; ++v) {
        #pragma unroll
        for (int off = 1; off < 16; off <<= 1) {
            psum[v] += __shfl_xor(psum[v], off, 32);
            pcnt[v] += __shfl_xor(pcnt[v], off, 32);
            nmin[v]  = fminf(nmin[v], __shfl_xor(nmin[v], off, 32));
        }
    }
    // One lane per half-wave folds into LDS (uint-min valid: d >= 0)
    if (ln == 0) {
        #pragma unroll
        for (int v = 0; v < 8; ++v) {
            const int rr = v + 8 * hi;      // 0..15 row slot in this block
            atomicAdd(&sPos[rr], psum[v]);
            atomicAdd(&sCnt[rr], pcnt[v]);
            atomicMin(&sMin[rr], __float_as_uint(nmin[v]));
        }
    }
    __syncthreads();

    if (t < 16) {
        const float pc = sCnt[t];
        const float nm = __uint_as_float(sMin[t]);
        const bool valid = (pc > 0.0f) && (nm < 2.9e38f); // neg exists
        if (valid) {
            const float posStat = sPos[t] / fmaxf(pc, 1.0f);
            const float per = fmaxf(posStat - nm + MARGIN, 0.0f);
            atomicAdd(&accum[0], per);
            atomicAdd(&accum[1], 1.0f);
        }
    }
}

// ---------------------------------------------------------------------------
// Kernel 3: finalize scalar mean over valid rows.
// ---------------------------------------------------------------------------
__global__ void tl_finalize(const float* __restrict__ accum,
                            float* __restrict__ out) {
    const float n = accum[1];
    out[0] = (n > 0.0f) ? (accum[0] / n) : 0.0f;
}

extern "C" void kernel_launch(void* const* d_in, const int* in_sizes, int n_in,
                              void* d_out, int out_size, void* d_ws, size_t ws_size,
                              hipStream_t stream) {
    const float*     emb    = (const float*)d_in[0];
    const long long* labels = (const long long*)d_in[1];

    char* ws = (char*)d_ws;
    _Float16* nf16 = (_Float16*)ws;
    float*    sqn  = (float*)(ws + WS_SQN_OFF);
    float*    acc  = (float*)(ws + WS_ACC_OFF);
    float*    out  = (float*)d_out;

    tl_normalize<<<NROWS, 128, 0, stream>>>(emb, nf16, sqn, acc);
    tl_gram_reduce<<<NROWS / 16, 256, 0, stream>>>(nf16, sqn, labels, acc);
    tl_finalize<<<1, 1, 0, stream>>>(acc, out);
}
